// ContrastiveLoss_20109036879978
// MI455X (gfx1250) — matrix operand order repaired and checked by reading, not compiled, not run on hardware
//
#include <hip/hip_runtime.h>

// Contrastive loss on MI455X (gfx1250).
//  - One-time f32 -> bf16 hi/lo split pass into d_ws (L2-resident, 16 MB).
//  - Gram matrix via 3x v_wmma_f32_16x16x32_bf16 per K=32 (hi*hi + hi*lo +
//    lo*hi ~ f32 accuracy), 128x128 LDS-staged tiles, register-pipelined
//    global->LDS staging (struct-by-value, SROA-friendly: no lambda/array
//    allocas -> no scratch traffic), upper triangle only (x2 weight).
//  - Branchless fused hinge/mask epilogue + wave/block reduce + one
//    global_atomic_add_f32 per block.

#define N_ROWS 8192
#define N_COLS 512
#define TB     128
#define MARGIN 1.0f

typedef __attribute__((ext_vector_type(16))) __bf16 v16bf;
typedef __attribute__((ext_vector_type(8)))  float  v8f;

struct Q2 { uint4 a, b; };   // 32 bytes == one v16bf fragment

// staging buffers held as individual struct members -> promoted to VGPRs
struct Stage {
  uint4 ah0, ah1, al0, al1, bh0, bh1, bl0, bl1;
};

static __device__ __forceinline__ unsigned short bf16_rne(float x) {
  unsigned u = __float_as_uint(x);
  u += 0x7FFFu + ((u >> 16) & 1u);        // round-to-nearest-even to bf16
  return (unsigned short)(u >> 16);
}

static __device__ __forceinline__ void split_pair(float x0, float x1,
                                                  unsigned &ph, unsigned &pl) {
  unsigned short h0 = bf16_rne(x0);
  unsigned short h1 = bf16_rne(x1);
  float r0 = x0 - __uint_as_float((unsigned)h0 << 16);
  float r1 = x1 - __uint_as_float((unsigned)h1 << 16);
  unsigned short l0 = bf16_rne(r0);
  unsigned short l1 = bf16_rne(r1);
  ph = (unsigned)h0 | ((unsigned)h1 << 16);
  pl = (unsigned)l0 | ((unsigned)l1 << 16);
}

// Fallback path helper: convert 16 f32 -> bf16 hi/lo, store to LDS.
static __device__ __forceinline__ void stage16(const float* __restrict__ g,
                                               unsigned short* hi_dst,
                                               unsigned short* lo_dst) {
  float4 f0 = ((const float4*)g)[0];
  float4 f1 = ((const float4*)g)[1];
  float4 f2 = ((const float4*)g)[2];
  float4 f3 = ((const float4*)g)[3];
  unsigned ph[8], pl[8];
  split_pair(f0.x, f0.y, ph[0], pl[0]);
  split_pair(f0.z, f0.w, ph[1], pl[1]);
  split_pair(f1.x, f1.y, ph[2], pl[2]);
  split_pair(f1.z, f1.w, ph[3], pl[3]);
  split_pair(f2.x, f2.y, ph[4], pl[4]);
  split_pair(f2.z, f2.w, ph[5], pl[5]);
  split_pair(f3.x, f3.y, ph[6], pl[6]);
  split_pair(f3.z, f3.w, ph[7], pl[7]);
  ((uint4*)hi_dst)[0] = make_uint4(ph[0], ph[1], ph[2], ph[3]);
  ((uint4*)hi_dst)[1] = make_uint4(ph[4], ph[5], ph[6], ph[7]);
  ((uint4*)lo_dst)[0] = make_uint4(pl[0], pl[1], pl[2], pl[3]);
  ((uint4*)lo_dst)[1] = make_uint4(pl[4], pl[5], pl[6], pl[7]);
}

// A fragment (16x32, 16-bit): elems 0-7 = K[ka..ka+7], elems 8-15 = K[16+ka..]
static __device__ __forceinline__ v16bf load_fragA(const unsigned short* row, int ka) {
  Q2 t;
  t.a = *(const uint4*)(row + ka);
  t.b = *(const uint4*)(row + 16 + ka);
  return __builtin_bit_cast(v16bf, t);
}
// B fragment (32x16): elems 0-7 = K[kb..kb+7], elems 8-15 = K[16+kb..]
static __device__ __forceinline__ v16bf load_fragB(const unsigned short* row, int kb) {
  Q2 t;
  t.a = *(const uint4*)(row + kb);
  t.b = *(const uint4*)(row + 16 + kb);
  return __builtin_bit_cast(v16bf, t);
}

static __device__ __forceinline__ Stage gload(const unsigned short* __restrict__ Xhi,
                                              const unsigned short* __restrict__ Xlo,
                                              size_t offA, size_t offB) {
  Stage s;
  const uint4* pah = (const uint4*)(Xhi + offA);
  const uint4* pal = (const uint4*)(Xlo + offA);
  const uint4* pbh = (const uint4*)(Xhi + offB);
  const uint4* pbl = (const uint4*)(Xlo + offB);
  s.ah0 = pah[0]; s.ah1 = pah[1];
  s.al0 = pal[0]; s.al1 = pal[1];
  s.bh0 = pbh[0]; s.bh1 = pbh[1];
  s.bl0 = pbl[0]; s.bl1 = pbl[1];
  return s;
}

// ---------------- prep 1: row squared norms (exact f32) + zero output -------
__global__ __launch_bounds__(128) void cl_prep_norm(const float* __restrict__ X,
                                                    float* __restrict__ sq,
                                                    float* __restrict__ out) {
  int row = blockIdx.x;
  int tid = threadIdx.x;                       // 128 threads, 4 floats each
  float4 v = ((const float4*)(X + (size_t)row * N_COLS))[tid];
  float s = v.x * v.x + v.y * v.y + v.z * v.z + v.w * v.w;
  for (int o = 16; o > 0; o >>= 1) s += __shfl_down(s, o, 32);
  __shared__ float part[4];
  if ((tid & 31) == 0) part[tid >> 5] = s;
  __syncthreads();
  if (tid == 0) {
    sq[row] = part[0] + part[1] + part[2] + part[3];
    if (row == 0) out[0] = 0.0f;               // harness poisons d_out
  }
}

// ---------------- prep 2: one-time f32 -> bf16 hi/lo split ------------------
__global__ __launch_bounds__(256) void cl_split(const float* __restrict__ X,
                                                unsigned short* __restrict__ Xhi,
                                                unsigned short* __restrict__ Xlo) {
  size_t t = (size_t)blockIdx.x * 256 + threadIdx.x;   // 8 floats per thread
  const float4* p = (const float4*)X + 2 * t;
  float4 f0 = p[0], f1 = p[1];
  unsigned ph[4], pl[4];
  split_pair(f0.x, f0.y, ph[0], pl[0]);
  split_pair(f0.z, f0.w, ph[1], pl[1]);
  split_pair(f1.x, f1.y, ph[2], pl[2]);
  split_pair(f1.z, f1.w, ph[3], pl[3]);
  ((uint4*)Xhi)[t] = make_uint4(ph[0], ph[1], ph[2], ph[3]);
  ((uint4*)Xlo)[t] = make_uint4(pl[0], pl[1], pl[2], pl[3]);
}

// ---------------- main: tiled WMMA Gram + fused loss epilogue ----------------
template <bool PRECONV>
__global__ __launch_bounds__(256, 2) void cl_main(const float* __restrict__ X,
                                                  const unsigned short* __restrict__ Xhi,
                                                  const unsigned short* __restrict__ Xlo,
                                                  const int* __restrict__ tgt,
                                                  const float* __restrict__ sq,
                                                  float* __restrict__ out) {
  const int bi = blockIdx.y;                   // row block
  const int bj = blockIdx.x;                   // col block
  if (bj < bi) return;                         // symmetry: upper triangle only
  const int i0 = bi * TB, j0 = bj * TB;

  const int tid  = threadIdx.x;
  const int lane = tid & 31;
  const int w    = tid >> 5;                   // wave id 0..7
  const int wm   = w >> 1;                     // 0..3 -> 32-row patch
  const int wn   = w & 1;                      // 0..1 -> 64-col patch
  const int hl   = lane >> 4;                  // lane half
  const int l15  = lane & 15;
  const int ka   = hl * 8;                     // A fragment K offset
  const int kb   = hl * 8;                     // B fragment K offset

  // padded rows: 40 ushorts = 80 B stride (16B aligned, bank-staggered)
  __shared__ unsigned short sAhi[TB][40], sAlo[TB][40];
  __shared__ unsigned short sBhi[TB][40], sBlo[TB][40];
  __shared__ float red[8];

  v8f acc[2][4] = {};

  const int r  = tid >> 1;                     // staging row 0..127
  const int ch = (tid & 1) << 4;               // staging col 0 or 16
  const size_t offA0 = (size_t)(i0 + r) * N_COLS + ch;
  const size_t offB0 = (size_t)(j0 + r) * N_COLS + ch;

  Stage st;
  if constexpr (PRECONV) st = gload(Xhi, Xlo, offA0, offB0);

  for (int k0 = 0; k0 < N_COLS; k0 += 32) {
    if constexpr (PRECONV) {
      ((uint4*)&sAhi[r][ch])[0] = st.ah0; ((uint4*)&sAhi[r][ch])[1] = st.ah1;
      ((uint4*)&sAlo[r][ch])[0] = st.al0; ((uint4*)&sAlo[r][ch])[1] = st.al1;
      ((uint4*)&sBhi[r][ch])[0] = st.bh0; ((uint4*)&sBhi[r][ch])[1] = st.bh1;
      ((uint4*)&sBlo[r][ch])[0] = st.bl0; ((uint4*)&sBlo[r][ch])[1] = st.bl1;
      // ds_store reads sources at issue -> safe to refill now; these global
      // loads fly during the barrier + fragment loads + WMMA below.
      if (k0 + 32 < N_COLS)
        st = gload(Xhi, Xlo, offA0 + k0 + 32, offB0 + k0 + 32);
    } else {
      stage16(X + offA0 + k0, &sAhi[r][ch], &sAlo[r][ch]);
      stage16(X + offB0 + k0, &sBhi[r][ch], &sBlo[r][ch]);
    }
    __syncthreads();

    v16bf ahi[2], alo[2];
#pragma unroll
    for (int mt = 0; mt < 2; ++mt) {
      int rr = wm * 32 + mt * 16 + l15;
      ahi[mt] = load_fragA(&sAhi[rr][0], ka);
      alo[mt] = load_fragA(&sAlo[rr][0], ka);
    }
    v16bf bhi[4], blo[4];
#pragma unroll
    for (int nt = 0; nt < 4; ++nt) {
      int cc = wn * 64 + nt * 16 + l15;
      bhi[nt] = load_fragB(&sBhi[cc][0], kb);
      blo[nt] = load_fragB(&sBlo[cc][0], kb);
    }

#pragma unroll
    for (int mt = 0; mt < 2; ++mt) {
#pragma unroll
      for (int nt = 0; nt < 4; ++nt) {
        v8f c = acc[mt][nt];
        c = __builtin_amdgcn_wmma_f32_16x16x32_bf16(false, ahi[mt], false, bhi[nt],
                                                    (short)0, c, false, false);
        c = __builtin_amdgcn_wmma_f32_16x16x32_bf16(false, ahi[mt], false, blo[nt],
                                                    (short)0, c, false, false);
        c = __builtin_amdgcn_wmma_f32_16x16x32_bf16(false, alo[mt], false, bhi[nt],
                                                    (short)0, c, false, false);
        acc[mt][nt] = c;
      }
    }
    __syncthreads();
  }

  // ---- branchless fused loss epilogue ----
  // C/D layout: lane holds N = lane&15; VGPR v holds M = v + 8*(lane>=16).
  float sqj[4];
  int   tjv[4];
#pragma unroll
  for (int nt = 0; nt < 4; ++nt) {
    int j = j0 + wn * 64 + nt * 16 + l15;
    sqj[nt] = sq[j];
    tjv[nt] = tgt[j];
  }

  float local = 0.0f;
#pragma unroll
  for (int mt = 0; mt < 2; ++mt) {
#pragma unroll
    for (int v = 0; v < 8; ++v) {
      int   i   = i0 + wm * 32 + mt * 16 + v + hl * 8;
      float sqi = sq[i];
      int   ti  = tgt[i];
#pragma unroll
      for (int nt = 0; nt < 4; ++nt) {
        int   j = j0 + wn * 64 + nt * 16 + l15;
        float d = fmaxf(sqi + sqj[nt] - 2.0f * acc[mt][nt][v], 0.0f);
        // d >= 0 so sqrtf never NaNs; max(M - sqrt(d), 0) == reference hinge
        float t = fmaxf(MARGIN - sqrtf(d), 0.0f);
        float c = (ti == tjv[nt]) ? d : t * t;
        local += (i != j) ? c : 0.0f;
      }
    }
  }

  const float inv2t = (float)(1.0 / (2.0 * (double)N_ROWS * (double)(N_ROWS - 1)));
  local *= inv2t * ((bj > bi) ? 2.0f : 1.0f);   // symmetry weight

  for (int o = 16; o > 0; o >>= 1) local += __shfl_down(local, o, 32);
  if (lane == 0) red[w] = local;
  __syncthreads();
  if (tid == 0) {
    float s = 0.0f;
#pragma unroll
    for (int q = 0; q < 8; ++q) s += red[q];
    atomicAdd(out, s);
  }
}

extern "C" void kernel_launch(void* const* d_in, const int* in_sizes, int n_in,
                              void* d_out, int out_size, void* d_ws, size_t ws_size,
                              hipStream_t stream) {
  const float* X   = (const float*)d_in[0];
  const int*   tgt = (const int*)d_in[1];
  float*       out = (float*)d_out;

  float*          sq  = (float*)d_ws;                              // 32 KB
  unsigned short* Xhi = (unsigned short*)((char*)d_ws + 32 * 1024);
  unsigned short* Xlo = Xhi + (size_t)N_ROWS * N_COLS;
  const size_t needed = 32 * 1024 + 2 * sizeof(unsigned short) * (size_t)N_ROWS * N_COLS;

  cl_prep_norm<<<dim3(N_ROWS), dim3(128), 0, stream>>>(X, sq, out);

  dim3 grid(N_ROWS / TB, N_ROWS / TB);
  if (ws_size >= needed) {
    cl_split<<<dim3((N_ROWS * N_COLS) / (256 * 8)), dim3(256), 0, stream>>>(X, Xhi, Xlo);
    cl_main<true><<<grid, 256, 0, stream>>>(X, Xhi, Xlo, tgt, sq, out);
  } else {
    cl_main<false><<<grid, 256, 0, stream>>>(X, nullptr, nullptr, tgt, sq, out);
  }
}
